// MoDE_52140902973544
// MI455X (gfx1250) — matrix operand order repaired and checked by reading, not compiled, not run on hardware
//
#include <hip/hip_runtime.h>
#include <hip/hip_bf16.h>

typedef __bf16 bf16_t;
typedef bf16_t v16bf __attribute__((ext_vector_type(16)));
typedef float  v8f   __attribute__((ext_vector_type(8)));
typedef int    i32x4 __attribute__((__vector_size__(16)));

#define NEXP 8
#define NE   7
#define Hh   1024
#define Ii   4096
#define NTOK 8192
#define CAP  1024

#define BM 128
#define BN 128
#define BK 32

#if defined(__has_builtin)
#if __has_builtin(__builtin_amdgcn_global_load_async_to_lds_b128) && \
    __has_builtin(__builtin_amdgcn_s_wait_asynccnt)
#define HAVE_ASYNC 1
#endif
#endif
#ifndef HAVE_ASYNC
#define HAVE_ASYNC 0
#endif

#define AS1 __attribute__((address_space(1)))
#define AS3 __attribute__((address_space(3)))

__device__ __forceinline__ void async_copy_b128(const void* g, void* l) {
#if HAVE_ASYNC
    __builtin_amdgcn_global_load_async_to_lds_b128((AS1 i32x4*)g, (AS3 i32x4*)l, 0, 0);
#else
    *(uint4*)l = *(const uint4*)g;
#endif
}

__device__ __forceinline__ void wait_async() {
#if HAVE_ASYNC
    __builtin_amdgcn_s_wait_asynccnt(0);
#endif
}

__device__ __forceinline__ unsigned short f2bf(float f) {
    union { float f; unsigned u; } v; v.f = f;
    unsigned r = v.u + 0x7FFFu + ((v.u >> 16) & 1u);   // round-to-nearest-even
    return (unsigned short)(r >> 16);
}

// ---------------- Kernel 1: router + softmax + top-2 ----------------
__global__ __launch_bounds__(32) void router_topk(
    const float* __restrict__ x, const float* __restrict__ rw,
    const float* __restrict__ rb, float* __restrict__ expert_w,
    float* __restrict__ noop_scale) {
    const int t = blockIdx.x;
    const int lane = threadIdx.x;
    const float4* xr = (const float4*)(x + (size_t)t * Hh);
    float acc[NEXP];
#pragma unroll
    for (int e = 0; e < NEXP; e++) acc[e] = 0.f;
#pragma unroll
    for (int j = 0; j < Hh / 4 / 32; j++) {
        float4 xv = xr[lane + 32 * j];
#pragma unroll
        for (int e = 0; e < NEXP; e++) {
            float4 wv = ((const float4*)(rw + (size_t)e * Hh))[lane + 32 * j];
            acc[e] += xv.x * wv.x + xv.y * wv.y + xv.z * wv.z + xv.w * wv.w;
        }
    }
#pragma unroll
    for (int off = 16; off > 0; off >>= 1) {
#pragma unroll
        for (int e = 0; e < NEXP; e++) acc[e] += __shfl_down(acc[e], off, 32);
    }
    if (lane == 0) {
        float lg[NEXP], w[NEXP];
        float mx = -1e30f;
#pragma unroll
        for (int e = 0; e < NEXP; e++) { lg[e] = acc[e] + rb[e]; mx = fmaxf(mx, lg[e]); }
        float sum = 0.f;
#pragma unroll
        for (int e = 0; e < NEXP; e++) { w[e] = __expf(lg[e] - mx); sum += w[e]; }
        const float inv = 1.f / sum;
        int i0 = 0;
#pragma unroll
        for (int e = 1; e < NEXP; e++) if (w[e] > w[i0]) i0 = e;
        int i1 = (i0 == 0) ? 1 : 0;
#pragma unroll
        for (int e = 0; e < NEXP; e++) if (e != i0 && w[e] > w[i1]) i1 = e;
#pragma unroll
        for (int e = 0; e < NE; e++)
            expert_w[(size_t)t * NE + e] = ((e == i0 || e == i1) ? w[e] * inv : 0.f);
        noop_scale[t] = ((i0 == NEXP - 1 || i1 == NEXP - 1) ? w[NEXP - 1] * inv : 0.f);
    }
}

// ------- Kernel 2: stable capacity assignment (ballot prefix scan) -------
__global__ __launch_bounds__(32) void assign_slots(
    const float* __restrict__ expert_w, int* __restrict__ slot_token,
    int* __restrict__ tok2slot) {
    const int e = blockIdx.x;
    const int lane = threadIdx.x;
    unsigned base = 0;
    for (int t0 = 0; t0 < NTOK; t0 += 32) {
        const int t = t0 + lane;
        const float w = expert_w[(size_t)t * NE + e];
        const bool routed = (w > 0.f);
        const unsigned mask = (unsigned)__ballot(routed);
        const unsigned pre = __popc(mask & ((1u << lane) - 1u));
        const int slot = (int)(base + pre);
        const bool valid = routed && (slot < CAP);
        tok2slot[(size_t)t * NE + e] = valid ? slot : -1;
        if (valid) slot_token[e * CAP + slot] = t;
        base += __popc(mask);
    }
    for (unsigned s = base + lane; s < CAP; s += 32) slot_token[e * CAP + s] = -1;
}

// ---------------- Kernel 3: gather routed tokens, fp32 -> bf16 ----------------
__global__ __launch_bounds__(128) void gather_tokens(
    const float* __restrict__ x, const int* __restrict__ slot_token,
    unsigned short* __restrict__ Abf) {
    const int e = blockIdx.y, s = blockIdx.x, tid = threadIdx.x;
    const int tok = slot_token[e * CAP + s];
    unsigned short* dst = Abf + ((size_t)e * CAP + s) * Hh;
    if (tok >= 0) {
        const float4* src = (const float4*)(x + (size_t)tok * Hh);
        for (int j = tid; j < Hh / 4; j += 128) {
            float4 v = src[j];
            uint2 p;
            p.x = (unsigned)f2bf(v.x) | ((unsigned)f2bf(v.y) << 16);
            p.y = (unsigned)f2bf(v.z) | ((unsigned)f2bf(v.w) << 16);
            *(uint2*)(dst + 4 * j) = p;
        }
    } else {
        uint2 z; z.x = 0u; z.y = 0u;
        for (int j = tid; j < Hh / 4; j += 128) *(uint2*)(dst + 4 * j) = z;
    }
}

// ------ Kernel 4: weight convert+transpose: Wt[n][k] = bf16(W[k][n]) ------
// W: per-expert (K x N) fp32 row-major; Wt: per-expert (N x K) bf16 row-major
__global__ __launch_bounds__(256) void convert_transpose(
    const float* __restrict__ W, unsigned short* __restrict__ Wt,
    const int K, const int N) {
    __shared__ unsigned short T[64][72];   // 72: rows stay 16B-aligned, no 2^n stride
    const int e = blockIdx.z;
    const int kb = blockIdx.y * 64;
    const int nb = blockIdx.x * 64;
    const float* Wsrc = W + (size_t)e * K * N;
    unsigned short* Wd = Wt + (size_t)e * N * K;
    const int tid = threadIdx.x;
#pragma unroll
    for (int rr = 0; rr < 64; rr += 16) {
        const int r = rr + (tid >> 4);
        const int c = (tid & 15) * 4;
        float4 f = *(const float4*)(Wsrc + (size_t)(kb + r) * N + nb + c);
        T[c + 0][r] = f2bf(f.x);
        T[c + 1][r] = f2bf(f.y);
        T[c + 2][r] = f2bf(f.z);
        T[c + 3][r] = f2bf(f.w);
    }
    __syncthreads();
#pragma unroll
    for (int w = 0; w < 2; w++) {
        const int idx = tid + w * 256;     // uint4 index; 8 uint4 per 64-elem row
        const int n = idx >> 3;
        const int kc = (idx & 7) * 8;
        *(uint4*)(Wd + (size_t)(nb + n) * K + kb + kc) = *(const uint4*)&T[n][kc];
    }
}

// ------ Kernel 5: unified expert GEMM, async double-buffered LDS, bf16 WMMA ------
// C[m][n] = sum_k A[m][k] * Bt[n][k];  A: (CAP x K) bf16, Bt: (N x K) bf16
template <bool RELU>
__global__ __launch_bounds__(256) void expert_gemm(
    const unsigned short* __restrict__ Abase,
    const unsigned short* __restrict__ Btbase,
    const int K, const int N,
    unsigned short* __restrict__ outBf, float* __restrict__ outF) {
    __shared__ unsigned short As[2][BM][BK];
    __shared__ unsigned short Bs[2][BN][BK];
    const int e = blockIdx.z;
    const int m0 = blockIdx.y * BM;
    const int n0 = blockIdx.x * BN;
    const unsigned short* A  = Abase + (size_t)e * CAP * K;
    const unsigned short* Bt = Btbase + (size_t)e * (size_t)N * K;
    const int tid = threadIdx.x;
    const int wave = tid >> 5, lane = tid & 31;
    const int wm = (wave & 3) * 32;        // 4 wave-rows of 32
    const int wn = (wave >> 2) * 64;       // 2 wave-cols of 64
    const int lhalf = lane >> 4, lmod = lane & 15;
    const int crow = tid >> 1;             // copy mapping: 2 threads/row, 32B each
    const int chalf = (tid & 1) * 16;

    v8f acc[2][4];
#pragma unroll
    for (int mi = 0; mi < 2; mi++)
#pragma unroll
        for (int ni = 0; ni < 4; ni++)
#pragma unroll
            for (int k = 0; k < 8; k++) acc[mi][ni][k] = 0.f;

    auto issue_tile = [&](int buf, int k0) {
        const unsigned short* ga = A + (size_t)(m0 + crow) * K + k0 + chalf;
        async_copy_b128(ga,     &As[buf][crow][chalf]);
        async_copy_b128(ga + 8, &As[buf][crow][chalf + 8]);
        const unsigned short* gb = Bt + (size_t)(n0 + crow) * K + k0 + chalf;
        async_copy_b128(gb,     &Bs[buf][crow][chalf]);
        async_copy_b128(gb + 8, &Bs[buf][crow][chalf + 8]);
    };

    issue_tile(0, 0);
    const int T = K / BK;
    for (int i = 0; i < T; i++) {
        const int buf = i & 1;
        wait_async();          // this wave's async copies for tile i are done
        __syncthreads();       // everyone done reading buf^1 (tile i-1) and filling buf
        if (i + 1 < T) issue_tile(buf ^ 1, (i + 1) * BK);

        union Fr { v16bf v; uint4 q[2]; };
        Fr af[2], bfr[4];
#pragma unroll
        for (int mi = 0; mi < 2; mi++) {   // A 16x32: lane<16 K{0..7,16..23}; lane>=16 K{8..15,24..31}
            const int r = wm + mi * 16 + lmod;
            const int ko = lhalf * 8;
            af[mi].q[0] = *(const uint4*)&As[buf][r][ko];
            af[mi].q[1] = *(const uint4*)&As[buf][r][16 + ko];
        }
#pragma unroll
        for (int ni = 0; ni < 4; ni++) {   // B 32x16: lane<16 K{0..7,16..23}; lane>=16 K{8..15,24..31}
            const int n = wn + ni * 16 + lmod;
            const int kb = lhalf * 8;
            bfr[ni].q[0] = *(const uint4*)&Bs[buf][n][kb];
            bfr[ni].q[1] = *(const uint4*)&Bs[buf][n][16 + kb];
        }
#pragma unroll
        for (int mi = 0; mi < 2; mi++)
#pragma unroll
            for (int ni = 0; ni < 4; ni++)
                acc[mi][ni] = __builtin_amdgcn_wmma_f32_16x16x32_bf16(
                    false, af[mi].v, false, bfr[ni].v, (short)0, acc[mi][ni], false, false);
    }

    // epilogue; C layout: n = lmod, m = r + 8*lhalf
    const int rowadd = lhalf * 8;
#pragma unroll
    for (int mi = 0; mi < 2; mi++)
#pragma unroll
        for (int ni = 0; ni < 4; ni++) {
            const int n = n0 + wn + ni * 16 + lmod;
#pragma unroll
            for (int r = 0; r < 8; r++) {
                const int m = m0 + wm + mi * 16 + r + rowadd;
                float v = acc[mi][ni][r];
                if (RELU) {
                    v = v > 0.f ? v : 0.f;
                    outBf[((size_t)e * CAP + m) * N + n] = f2bf(v);
                } else {
                    outF[((size_t)e * CAP + m) * N + n] = v;
                }
            }
        }
}

// ------- Kernel 6: deterministic combine: noop path + <=2 expert contributions -------
__global__ __launch_bounds__(256) void combine_out(
    const float* __restrict__ x, const float* __restrict__ noop_scale,
    const float* __restrict__ expert_w, const int* __restrict__ tok2slot,
    const float* __restrict__ Eout, float* __restrict__ out) {
    const int t = blockIdx.x;
    const int tid = threadIdx.x;
    const float s = noop_scale[t];
    float4 v = ((const float4*)(x + (size_t)t * Hh))[tid];
    float4 a;
    a.x = v.x * s; a.y = v.y * s; a.z = v.z * s; a.w = v.w * s;
#pragma unroll
    for (int e = 0; e < NE; e++) {
        const float w = expert_w[(size_t)t * NE + e];
        const int slot = tok2slot[(size_t)t * NE + e];
        if (w > 0.f && slot >= 0) {
            float4 u = ((const float4*)(Eout + ((size_t)e * CAP + slot) * Hh))[tid];
            a.x += u.x * w; a.y += u.y * w; a.z += u.z * w; a.w += u.w * w;
        }
    }
    ((float4*)(out + (size_t)t * Hh))[tid] = a;
}

extern "C" void kernel_launch(void* const* d_in, const int* in_sizes, int n_in,
                              void* d_out, int out_size, void* d_ws, size_t ws_size,
                              hipStream_t stream) {
    const float* x  = (const float*)d_in[0];
    const float* rw = (const float*)d_in[1];
    const float* rb = (const float*)d_in[2];
    const float* W1 = (const float*)d_in[3];
    const float* W2 = (const float*)d_in[4];
    float* out = (float*)d_out;
    (void)in_sizes; (void)n_in; (void)out_size; (void)ws_size;

    char* ws = (char*)d_ws;
    size_t off = 0;
    auto alloc = [&](size_t bytes) -> char* {
        char* p = ws + off;
        off = (off + bytes + 255) & ~(size_t)255;
        return p;
    };
    float*          expert_w   = (float*)alloc((size_t)NTOK * NE * 4);
    float*          noop_scale = (float*)alloc((size_t)NTOK * 4);
    int*            slot_token = (int*)alloc((size_t)NE * CAP * 4);
    int*            tok2slot   = (int*)alloc((size_t)NTOK * NE * 4);
    unsigned short* Abf        = (unsigned short*)alloc((size_t)NE * CAP * Hh * 2);
    unsigned short* Hact       = (unsigned short*)alloc((size_t)NE * CAP * Ii * 2);
    float*          Eout       = (float*)alloc((size_t)NE * CAP * Hh * 4);
    unsigned short* Wt         = (unsigned short*)alloc((size_t)NE * Hh * Ii * 2); // reused W1t/W2t

    router_topk<<<NTOK, 32, 0, stream>>>(x, rw, rb, expert_w, noop_scale);
    assign_slots<<<NE, 32, 0, stream>>>(expert_w, slot_token, tok2slot);
    gather_tokens<<<dim3(CAP, NE), 128, 0, stream>>>(x, slot_token, Abf);

    // FFN layer 1: Wt = W1^T (bf16, L2-resident), h = relu(A @ W1)
    convert_transpose<<<dim3(Ii / 64, Hh / 64, NE), 256, 0, stream>>>(W1, Wt, Hh, Ii);
    expert_gemm<true><<<dim3(Ii / BN, CAP / BM, NE), 256, 0, stream>>>(
        Abf, Wt, Hh, Ii, Hact, nullptr);

    // FFN layer 2: Wt = W2^T (bf16, reuse buffer), Eout = h @ W2
    convert_transpose<<<dim3(Hh / 64, Ii / 64, NE), 256, 0, stream>>>(W2, Wt, Ii, Hh);
    expert_gemm<false><<<dim3(Hh / BN, CAP / BM, NE), 256, 0, stream>>>(
        Hact, Wt, Ii, Hh, nullptr, Eout);

    combine_out<<<NTOK, 256, 0, stream>>>(x, noop_scale, expert_w, tok2slot, Eout, out);
}
